// Associative3DEmbedding_87351044866374
// MI455X (gfx1250) — matrix operand order repaired and checked by reading, not compiled, not run on hardware
//
#include <hip/hip_runtime.h>
#include <hip/hip_bf16.h>
#include <math.h>

typedef __attribute__((ext_vector_type(16))) _Float16 v16h;
typedef __attribute__((ext_vector_type(8)))  _Float16 v8h;
typedef __attribute__((ext_vector_type(4)))  _Float16 v4h;
typedef __attribute__((ext_vector_type(8)))  float    v8f;

#define NZ   2048
#define H1   256
#define H2   128
#define H3   64
#define EPS  1e-8f

// LDS row strides padded +8 halves (+16B) so row stride mod 256B == 16B
// -> 16 A-fragment rows land on banks {0,4,...,60}: conflict-free ds_load_b128.
#define LDF  (NZ + 8)    // featbuf stride (halves)
#define LD1  (H1 + 8)    // hbuf1 stride (halves)
#define LD2  (H2 + 8)    // hbuf2 stride (halves)
#define LDE  (H3 + 1)    // ebuf stride (floats)

// ---------------------------------------------------------------------------
// WMMA fragment builders (wave32, V_WMMA_F32_16X16X32_F16 layouts per ISA 7.12.2)
// ---------------------------------------------------------------------------

// A-matrix 16x32 f16 from a row-major f16 tile (LDS or global; stride ld halves).
// lanes 0-15: row=lane, elems 0-7 -> K=kb+0..7, elems 8-15 -> K=kb+16..23
// lanes16-31: row=lane-16, elems 0-7 -> K=kb+8..15, elems 8-15 -> K=kb+24..31
static __device__ __forceinline__ v16h a_frag_f16(const _Float16* __restrict__ base,
                                                  int ld, int kb) {
    const int lane = threadIdx.x & 31;
    const _Float16* p = base + (size_t)(lane & 15) * ld + kb + ((lane >> 4) << 3);
    v8h lo = *(const v8h*)p;
    v8h hi = *(const v8h*)(p + 16);
    return __builtin_shufflevector(lo, hi, 0, 1, 2, 3, 4, 5, 6, 7,
                                           8, 9, 10, 11, 12, 13, 14, 15);
}

// B-matrix 32x16 f16 from N-major f16 storage (wt[col][k], col stride ld halves).
// lanes 0-15: col=cb+lane,   elems 0-7 -> K=kb+0..7, elems 8-15 -> K=kb+16..23
// lanes16-31: col=cb+lane-16,elems 0-7 -> K=kb+8..15, elems 8-15 -> K=kb+24..31
static __device__ __forceinline__ v16h b_frag_f16(const _Float16* __restrict__ wt,
                                                  int ld, int cb, int kb) {
    const int lane = threadIdx.x & 31;
    const _Float16* p = wt + (size_t)(cb + (lane & 15)) * ld + kb + ((lane >> 4) << 3);
    v8h lo = *(const v8h*)p;
    v8h hi = *(const v8h*)(p + 16);
    return __builtin_shufflevector(lo, hi, 0, 1, 2, 3, 4, 5, 6, 7,
                                           8, 9, 10, 11, 12, 13, 14, 15);
}

static __device__ __forceinline__ v8f wmma_f16(v16h a, v16h b, v8f c) {
    return __builtin_amdgcn_wmma_f32_16x16x32_f16(false, a, false, b,
                                                  (short)0, c, false, false);
}

// ---------------------------------------------------------------------------
// Kernel 1: convert + transpose weights to f16 N-major
// ---------------------------------------------------------------------------
__global__ void prep_weights(const float* __restrict__ W1,
                             const float* __restrict__ W2,
                             const float* __restrict__ W3,
                             _Float16* __restrict__ W1t,   // [256][2048]
                             _Float16* __restrict__ W2t,   // [128][256]
                             _Float16* __restrict__ W3t) { // [64][128]
    int tid = blockIdx.x * blockDim.x + threadIdx.x;
    if (tid < NZ * H1) {                       // W1 [2048][256]
        int k = tid >> 8, n = tid & 255;
        W1t[(size_t)n * NZ + k] = (_Float16)W1[tid];
    }
    if (tid < H1 * H2) {                       // W2 [256][128]
        int k = tid >> 7, n = tid & 127;
        W2t[(size_t)n * H1 + k] = (_Float16)W2[tid];
    }
    if (tid < H2 * H3) {                       // W3 [128][64]
        int k = tid >> 6, n = tid & 63;
        W3t[(size_t)n * H2 + k] = (_Float16)W3[tid];
    }
}

// ---------------------------------------------------------------------------
// Kernel 2: fused 3-layer MLP + L2-normalize, 16 rows per block, 8 waves.
// feat tile is converted f32->f16 into LDS ONCE (no 8x redundant load+cvt).
// ---------------------------------------------------------------------------
__global__ __launch_bounds__(256) void mlp_embed(
    const float* __restrict__ feat,       // [nrows][2048]
    const _Float16* __restrict__ W1t,     // [256][2048]
    const float* __restrict__ b1,
    const _Float16* __restrict__ W2t,     // [128][256]
    const float* __restrict__ b2,
    const _Float16* __restrict__ W3t,     // [64][128]
    const float* __restrict__ b3,
    _Float16* __restrict__ nrm)           // [nrows][64] normalized, f16
{
    __shared__ __align__(32) _Float16 featbuf[16 * LDF]; // ~64 KB (A tile, f16)
    __shared__ __align__(32) _Float16 hbuf1[16 * LD1];   // ~8 KB (layer-1 out)
    __shared__ __align__(32) _Float16 hbuf2[16 * LD2];   // ~4 KB (layer-2 out)
    __shared__ __align__(32) float    ebuf[16 * LDE];    // ~4 KB (layer-3 out)
    __shared__ float rnorm[16];

    const int w    = threadIdx.x >> 5;
    const int lane = threadIdx.x & 31;
    const int colL = lane & 15;                 // column within a 16-wide tile
    const int rowO = (lane >> 4) << 3;          // C-layout: +8 rows for hi half-wave
    const float* arow = feat + (size_t)blockIdx.x * 16 * NZ;

    // ----- stage feat tile to LDS as f16 (one pass; 8-elem chunks, b128 ld/st)
    {
        // 16*2048/8 = 4096 chunks; 256 threads * 16 iters
#pragma unroll
        for (int it = 0; it < 16; ++it) {
            int c   = threadIdx.x + it * 256;
            int row = c >> 8;                  // 256 chunks per row
            int col = (c & 255) << 3;
            v8f x = *(const v8f*)(arow + (size_t)row * NZ + col);
            v8h h;
#pragma unroll
            for (int j = 0; j < 8; ++j) h[j] = (_Float16)x[j];
            *(v8h*)(featbuf + row * LDF + col) = h;
        }
    }
    __syncthreads();

    // ----- layer 1: [16 x 2048] x [2048 x 256], wave w owns cols [32w, 32w+32)
    {
        v8f c0 = {}, c1 = {};
        const int cb = w * 32;
#pragma unroll 4
        for (int kb = 0; kb < NZ; kb += 32) {
            v16h a  = a_frag_f16(featbuf, LDF, kb);
            v16h bA = b_frag_f16(W1t, NZ, cb, kb);
            v16h bB = b_frag_f16(W1t, NZ, cb + 16, kb);
            c0 = wmma_f16(a, bA, c0);
            c1 = wmma_f16(a, bB, c1);
        }
        const float biasA = b1[cb + colL];
        const float biasB = b1[cb + 16 + colL];
#pragma unroll
        for (int v = 0; v < 8; ++v) {
            int m = v + rowO;
            hbuf1[m * LD1 + cb + colL]      = (_Float16)fmaxf(c0[v] + biasA, 0.0f);
            hbuf1[m * LD1 + cb + 16 + colL] = (_Float16)fmaxf(c1[v] + biasB, 0.0f);
        }
    }
    __syncthreads();

    // ----- layer 2: [16 x 256] x [256 x 128], wave w owns cols [16w, 16w+16)
    {
        v8f c = {};
        const int cb = w * 16;
#pragma unroll
        for (int kb = 0; kb < H1; kb += 32) {
            v16h a = a_frag_f16(hbuf1, LD1, kb);
            v16h b = b_frag_f16(W2t, H1, cb, kb);
            c = wmma_f16(a, b, c);
        }
        const float bias = b2[cb + colL];
#pragma unroll
        for (int v = 0; v < 8; ++v)
            hbuf2[(v + rowO) * LD2 + cb + colL] = (_Float16)fmaxf(c[v] + bias, 0.0f);
    }
    __syncthreads();

    // ----- layer 3: [16 x 128] x [128 x 64], waves 0..3 (wave-uniform branch,
    //       EXEC stays all-ones inside WMMA)
    if (w < 4) {
        v8f c = {};
        const int cb = w * 16;
#pragma unroll
        for (int kb = 0; kb < H2; kb += 32) {
            v16h a = a_frag_f16(hbuf2, LD2, kb);
            v16h b = b_frag_f16(W3t, H2, cb, kb);
            c = wmma_f16(a, b, c);
        }
        const float bias = b3[cb + colL];
#pragma unroll
        for (int v = 0; v < 8; ++v)
            ebuf[(v + rowO) * LDE + cb + colL] = c[v] + bias;
    }
    __syncthreads();

    // ----- row norms (16 rows, one thread each; padded stride -> no conflicts)
    if (threadIdx.x < 16) {
        float s = 0.0f;
#pragma unroll 8
        for (int j = 0; j < H3; ++j) {
            float v = ebuf[threadIdx.x * LDE + j];
            s += v * v;
        }
        rnorm[threadIdx.x] = 1.0f / fmaxf(sqrtf(s), EPS);
    }
    __syncthreads();

    // ----- normalize + store f16 (4 halves per thread, 8B stores)
    {
        int idx = threadIdx.x * 4;            // 256 threads * 4 = 1024 = 16*64
        int row = idx >> 6;
        int col = idx & 63;
        float r = rnorm[row];
        v4h o;
#pragma unroll
        for (int i = 0; i < 4; ++i) o[i] = (_Float16)(ebuf[row * LDE + col + i] * r);
        *(v4h*)(nrm + (size_t)blockIdx.x * 16 * H3 + idx) = o;
    }
}

// ---------------------------------------------------------------------------
// Kernel 3: out = sigmoid(5 * n1 @ n2^T), tiled WMMA, store-bound stream
// block = 8 waves; block tile 128 rows x 64 cols; wave tile 16 x 64
// ---------------------------------------------------------------------------
__global__ __launch_bounds__(256) void cosine_sigmoid(
    const _Float16* __restrict__ n1,   // [N1][64] row-major
    const _Float16* __restrict__ n2,   // [N2][64] row-major == N-major for B
    float* __restrict__ out, int N2)
{
    const int w     = threadIdx.x >> 5;
    const int lane  = threadIdx.x & 31;
    const int colL  = lane & 15;
    const int rowO  = (lane >> 4) << 3;
    const int mbase = blockIdx.y * 128 + w * 16;
    const int nbase = blockIdx.x * 64;

    // A fragments for K = 0..31 and 32..63 (reused across 4 N-tiles)
    const _Float16* abase = n1 + (size_t)mbase * H3;
    v16h a0 = a_frag_f16(abase, H3, 0);
    v16h a1 = a_frag_f16(abase, H3, 32);

    v8f c[4] = {{}, {}, {}, {}};
#pragma unroll
    for (int t = 0; t < 4; ++t) {
        v16h b0 = b_frag_f16(n2, H3, nbase + t * 16, 0);
        v16h b1 = b_frag_f16(n2, H3, nbase + t * 16, 32);
        c[t] = wmma_f16(a0, b0, c[t]);
        c[t] = wmma_f16(a1, b1, c[t]);
    }

    // fused sigmoid(5x) + coalesced f32 stores (16 lanes x 4B contiguous)
#pragma unroll
    for (int t = 0; t < 4; ++t) {
#pragma unroll
        for (int v = 0; v < 8; ++v) {
            float x = c[t][v];
            float s = 1.0f / (1.0f + __expf(-5.0f * x));
            size_t off = (size_t)(mbase + v + rowO) * N2 + nbase + t * 16 + colL;
            out[off] = s;
        }
    }
}

// ---------------------------------------------------------------------------
extern "C" void kernel_launch(void* const* d_in, const int* in_sizes, int n_in,
                              void* d_out, int out_size, void* d_ws, size_t ws_size,
                              hipStream_t stream) {
    const float* feat1 = (const float*)d_in[0];
    const float* feat2 = (const float*)d_in[1];
    const float* W1    = (const float*)d_in[2];
    const float* b1    = (const float*)d_in[3];
    const float* W2    = (const float*)d_in[4];
    const float* b2    = (const float*)d_in[5];
    const float* W3    = (const float*)d_in[6];
    const float* b3    = (const float*)d_in[7];
    float* out = (float*)d_out;

    const int N1 = in_sizes[0] / NZ;   // 8192
    const int N2 = in_sizes[1] / NZ;   // 8192

    // workspace layout (all 256B-aligned offsets)
    char* ws = (char*)d_ws;
    _Float16* W1t = (_Float16*)(ws);                                   // 1 MB
    _Float16* W2t = (_Float16*)(ws + (size_t)H1 * NZ * 2);             // 64 KB
    _Float16* W3t = (_Float16*)(ws + (size_t)H1 * NZ * 2 + H2 * H1 * 2);
    char* p = ws + (size_t)H1 * NZ * 2 + (size_t)H2 * H1 * 2 + (size_t)H3 * H2 * 2;
    p = (char*)(((uintptr_t)p + 255) & ~(uintptr_t)255);
    _Float16* nrm1 = (_Float16*)p;                                     // 1 MB
    _Float16* nrm2 = (_Float16*)(p + (size_t)N1 * H3 * 2);             // 1 MB

    // 1) weight transpose/convert
    prep_weights<<<(NZ * H1 + 255) / 256, 256, 0, stream>>>(W1, W2, W3, W1t, W2t, W3t);

    // 2) fused MLP + normalize for both feature sets
    mlp_embed<<<N1 / 16, 256, 0, stream>>>(feat1, W1t, b1, W2t, b2, W3t, b3, nrm1);
    mlp_embed<<<N2 / 16, 256, 0, stream>>>(feat2, W1t, b1, W2t, b2, W3t, b3, nrm2);

    // 3) cosine similarity + sigmoid (output-store bound)
    dim3 grid(N2 / 64, N1 / 128);
    cosine_sigmoid<<<grid, 256, 0, stream>>>(nrm1, nrm2, out, N2);
}
